// LSTMEncoder_66992899883552
// MI455X (gfx1250) — matrix operand order repaired and checked by reading, not compiled, not run on hardware
//
#include <hip/hip_runtime.h>
#include <hip/hip_bf16.h>
#include <stdint.h>

// ---------------------------------------------------------------------------
// LSTM encoder for MI455X (gfx1250), wave32 + WMMA bf16.
//   B=64, T=256, D=1024, U=1024, G=4*U=4096
//   1) xz = x@W + b : parallel WMMA GEMM (M=16384,K=1024,N=4096)
//   2) per t (serial): fused kernel: z = xz[:,t,:] + h@U (WMMA) -> LDS ->
//      gates + c/h update in the same block.
// K-loop: parity double-buffer + induction pointers with constant immediate
// offsets (no per-iteration 64-bit address math, no spills).
// W/U pre-packed into native 32x16 bf16 B-fragment layout (2 b128s per lane).
// ---------------------------------------------------------------------------

typedef __attribute__((ext_vector_type(16))) __bf16 v16bf;
typedef __attribute__((ext_vector_type(8)))  float  v8f;

#define LSTM_B   64
#define LSTM_T   256
#define LSTM_D   1024
#define LSTM_U   1024
#define LSTM_G   4096           // 4*U
#define KB_D     (LSTM_D / 32)  // 32 k-blocks of 32
#define NT_G     (LSTM_G / 16)  // 256 n-tiles of 16
#define B_KB_STRIDE   512       // halves per kb step inside packed tile
#define B_TILE_STRIDE 16384     // halves per n-tile in packed layout

union FragBF { uint4 u[2]; v16bf v; };

// A fragment: p = rowbase + 8*h (+ kb*32 via constant offset).
// halves [0..7] at p, halves [16..23] at p+16.
static __device__ inline v16bf load_a_frag(const __bf16* p) {
    FragBF f;
    f.u[0] = *(const uint4*)(p);
    f.u[1] = *(const uint4*)(p + 16);
    return f.v;
}

// Packed B fragment: 16 contiguous halves at p.
static __device__ inline v16bf load_b_frag(const __bf16* p) {
    FragBF f;
    f.u[0] = *(const uint4*)(p);
    f.u[1] = *(const uint4*)(p + 8);
    return f.v;
}

static __device__ inline v8f wmma_bf16(v16bf a, v16bf b, v8f c) {
    return __builtin_amdgcn_wmma_f32_16x16x32_bf16(false, a, false, b, (short)0, c,
                                                   false, false);
}

// Pipelined 16x64 strip GEMM over K=1024: acc[4] += A(16xK) * B(Kx64).
// aptr = A row base + 8*h; bptr = pack + (ntile0*KB_D*32 + lane)*16.
// NOTE: issues one redundant kb-block prefetch past the end (values unused);
// both source buffers carry >=4KB of padding in the workspace.
static __device__ inline void gemm_strip(const __bf16* __restrict__ aptr,
                                         const __bf16* __restrict__ bptr,
                                         v8f acc[4]) {
    v16bf a0, a1, b0[4], b1[4];
    a0 = load_a_frag(aptr);
    #pragma unroll
    for (int j = 0; j < 4; ++j) b0[j] = load_b_frag(bptr + j * B_TILE_STRIDE);

    for (int kb = 0; kb < KB_D; kb += 2) {
        // prefetch kb+1 into buffer 1; compute kb from buffer 0
        a1 = load_a_frag(aptr + 32);
        #pragma unroll
        for (int j = 0; j < 4; ++j)
            b1[j] = load_b_frag(bptr + B_KB_STRIDE + j * B_TILE_STRIDE);
        #pragma unroll
        for (int j = 0; j < 4; ++j) acc[j] = wmma_bf16(a0, b0[j], acc[j]);

        // prefetch kb+2 into buffer 0 (redundant at tail); compute kb+1
        a0 = load_a_frag(aptr + 64);
        #pragma unroll
        for (int j = 0; j < 4; ++j)
            b0[j] = load_b_frag(bptr + 2 * B_KB_STRIDE + j * B_TILE_STRIDE);
        #pragma unroll
        for (int j = 0; j < 4; ++j) acc[j] = wmma_bf16(a1, b1[j], acc[j]);

        aptr += 64;                 // 2 kb * 32 halves
        bptr += 2 * B_KB_STRIDE;    // 2 kb * 512 halves
    }
}

// --------------------------- setup kernels ---------------------------------

__global__ void cvt_f32_bf16_kernel(const float* __restrict__ src,
                                    __bf16* __restrict__ dst, int n) {
    int i = blockIdx.x * blockDim.x + threadIdx.x;
    if (i < n) dst[i] = (__bf16)src[i];
}

// Pack row-major [K, 4096] f32 into bf16 B-fragment layout.
__global__ void pack_b_kernel(const float* __restrict__ src,   // [K, 4096]
                              __bf16* __restrict__ pack) {
    int tid  = blockIdx.x * blockDim.x + threadIdx.x;     // NT_G*KB_D*32 threads
    int lane = tid & 31;
    int kb   = (tid >> 5) & (KB_D - 1);
    int tile = tid >> 10;
    int n    = tile * 16 + (lane & 15);
    int h    = lane >> 4;
    __bf16* dst = pack + (size_t)tid * 16;
    #pragma unroll
    for (int e = 0; e < 16; ++e) {
        int k = kb * 32 + (e < 8 ? 8 * h + e : 16 + 8 * h + (e - 8));
        dst[e] = (__bf16)src[(size_t)k * LSTM_G + n];
    }
}

__global__ void init_state_kernel(float* __restrict__ c, __bf16* __restrict__ hbf) {
    int i = blockIdx.x * blockDim.x + threadIdx.x;        // B*U threads
    c[i]   = 0.0f;
    hbf[i] = (__bf16)0.0f;
}

// --------------------------- big GEMM: xz = x@W + b -------------------------
// grid: (G/256, M/16), block: 128 (4 waves). Each wave: independent 16x64 strip.
__global__ void __launch_bounds__(128, 1)
gemm_xz_kernel(const __bf16* __restrict__ xbf,   // [16384,1024]
               const __bf16* __restrict__ wpack,
               const float*  __restrict__ bias,  // [4096]
               float* __restrict__ xz) {         // [16384,4096]
    const int lane   = threadIdx.x & 31;
    const int w      = threadIdx.x >> 5;
    const int mtile  = blockIdx.y;
    const int ntile0 = (blockIdx.x * 4 + w) * 4;
    const int r = lane & 15, h = lane >> 4;

    const __bf16* aptr = xbf + (size_t)(mtile * 16 + r) * LSTM_D + 8 * h;
    const __bf16* bptr = wpack + ((size_t)ntile0 * (KB_D * 32) + lane) * 16;

    v8f acc[4] = {};
    gemm_strip(aptr, bptr, acc);

    float* xzp = xz + (size_t)(mtile * 16 + 8 * h) * LSTM_G;
    #pragma unroll
    for (int j = 0; j < 4; ++j) {
        int col = (ntile0 + j) * 16 + r;
        float bv = bias[col];
        #pragma unroll
        for (int v = 0; v < 8; ++v)
            xzp[(size_t)v * LSTM_G + col] = acc[j][v] + bv;
    }
}

// ----------------- fused recurrent step: z-GEMM + gates ---------------------
// grid: (U/64=16, B/16=4), block: 128 (4 waves). Wave w computes gate w over
// the block's 16(b) x 64(u) patch; z staged in LDS; gates fused after barrier.
static __device__ inline float sigmoidf(float x) { return 1.0f / (1.0f + __expf(-x)); }

__global__ void __launch_bounds__(128, 1)
lstm_step_kernel(const __bf16* __restrict__ hbf_in,  // [64,1024]
                 const __bf16* __restrict__ upack,
                 const float*  __restrict__ xz,      // [16384,4096]
                 float* __restrict__ cbuf,           // [64,1024]
                 __bf16* __restrict__ hbf_out,       // [64,1024]
                 float* __restrict__ out,            // [B,T,U]
                 int t) {
    __shared__ float zl[4][16][64];                  // 16 KB: z per gate

    const int tid   = threadIdx.x;
    const int lane  = tid & 31;
    const int w     = tid >> 5;                      // wave id == gate id
    const int ut    = blockIdx.x;                    // u-strip (64 wide)
    const int mtile = blockIdx.y;                    // b-tile (16 rows)
    const int r = lane & 15, h = lane >> 4;

    const int ntile0 = w * 64 + ut * 4;              // tile index into G=4096
    const __bf16* aptr = hbf_in + (size_t)(mtile * 16 + r) * LSTM_U + 8 * h;
    const __bf16* bptr = upack + ((size_t)ntile0 * (KB_D * 32) + lane) * 16;

    v8f acc[4] = {};
    gemm_strip(aptr, bptr, acc);

    // z = h@U + xz[:,t,:]  -> LDS patch for this gate
    const float* xzp = xz + ((size_t)(mtile * 16 + 8 * h) * LSTM_T + t) * LSTM_G;
    #pragma unroll
    for (int j = 0; j < 4; ++j) {
        int col = (ntile0 + j) * 16 + r;             // column in G
        int lc  = j * 16 + r;                        // column in 64-wide patch
        #pragma unroll
        for (int v = 0; v < 8; ++v) {
            int row = 8 * h + v;                     // b within tile
            zl[w][row][lc] = acc[j][v] + xzp[(size_t)v * LSTM_T * LSTM_G + col];
        }
    }
    __syncthreads();

    // gates: 16x64 = 1024 (b,u) pairs, 128 threads x 8 each
    #pragma unroll
    for (int i = 0; i < 8; ++i) {
        int e   = tid + i * 128;
        int row = e >> 6;
        int col = e & 63;
        int b   = mtile * 16 + row;
        int u   = ut * 64 + col;
        float ig = sigmoidf(zl[0][row][col]);
        float fg = sigmoidf(zl[1][row][col]);
        float gg = tanhf(zl[2][row][col]);
        float og = sigmoidf(zl[3][row][col]);
        int ci = b * LSTM_U + u;
        float cn = fg * cbuf[ci] + ig * gg;
        cbuf[ci] = cn;
        float hn = og * tanhf(cn);
        out[((size_t)b * LSTM_T + t) * LSTM_U + u] = hn;
        hbf_out[ci] = (__bf16)hn;
    }
}

// ------------------------------- launcher -----------------------------------

extern "C" void kernel_launch(void* const* d_in, const int* in_sizes, int n_in,
                              void* d_out, int out_size, void* d_ws, size_t ws_size,
                              hipStream_t stream) {
    const float* x  = (const float*)d_in[0];   // [64,256,1024]
    const float* W  = (const float*)d_in[1];   // [1024,4096]
    const float* U  = (const float*)d_in[2];   // [1024,4096]
    const float* bb = (const float*)d_in[3];   // [4096]
    float* out = (float*)d_out;                // [64,256,1024]

    const size_t M   = (size_t)LSTM_B * LSTM_T;  // 16384
    const size_t PAD = 4096;                     // overrun slack per buffer

    // workspace carve-up (~305 MB total, each buffer padded for the
    // redundant tail prefetch in gemm_strip)
    char* p = (char*)d_ws;
    __bf16* xbf   = (__bf16*)p;  p += M * LSTM_D * sizeof(__bf16) + PAD;
    __bf16* wpack = (__bf16*)p;  p += (size_t)LSTM_D * LSTM_G * sizeof(__bf16) + PAD;
    __bf16* upack = (__bf16*)p;  p += (size_t)LSTM_U * LSTM_G * sizeof(__bf16) + PAD;
    float*  xz    = (float*)p;   p += M * LSTM_G * sizeof(float) + PAD;
    float*  cbuf  = (float*)p;   p += (size_t)LSTM_B * LSTM_U * sizeof(float) + PAD;
    __bf16* hbf   = (__bf16*)p;  p += (size_t)LSTM_B * LSTM_U * sizeof(__bf16) + PAD;
    (void)ws_size; (void)in_sizes; (void)n_in; (void)out_size;

    // 1) convert x to bf16
    {
        int n = (int)(M * LSTM_D);
        cvt_f32_bf16_kernel<<<dim3((n + 255) / 256), dim3(256), 0, stream>>>(x, xbf, n);
    }
    // 2) pack W, U into B-fragment layout
    {
        int n = NT_G * KB_D * 32;   // 262144 threads
        pack_b_kernel<<<dim3(n / 256), dim3(256), 0, stream>>>(W, wpack);
        pack_b_kernel<<<dim3(n / 256), dim3(256), 0, stream>>>(U, upack);
    }
    // 3) init recurrent state
    init_state_kernel<<<dim3((LSTM_B * LSTM_U) / 256), dim3(256), 0, stream>>>(cbuf, hbf);

    // 4) xz = x@W + b
    gemm_xz_kernel<<<dim3(LSTM_G / 256, (int)(M / 16)), dim3(128), 0, stream>>>(
        xbf, wpack, bb, xz);

    // 5) sequential scan over T: one fused kernel per step
    for (int t = 0; t < LSTM_T; ++t) {
        lstm_step_kernel<<<dim3(LSTM_U / 64, LSTM_B / 16), dim3(128), 0, stream>>>(
            hbf, upack, xz, cbuf, hbf, out, t);
    }
}
